// Head_23313082482790
// MI455X (gfx1250) — compile-verified, matrix operand order checked by reading, not run on hardware
//
#include <hip/hip_runtime.h>

typedef __attribute__((ext_vector_type(16))) __bf16 v16bf;
typedef __attribute__((ext_vector_type(8)))  float  v8f;

#define BATCH 8
#define TSEQ  4096
#define NEMB  128
#define HS    32
#define SCALE 0.17677669529663687f   // 1/sqrt(32), folded into q in phase 1

static __device__ __forceinline__ v8f wmma_bf16(v16bf a, v16bf b, v8f c) {
  // D = A(16x32 bf16) * B(32x16 bf16) + C(16x16 f32)
  return __builtin_amdgcn_wmma_f32_16x16x32_bf16(
      false, a, false, b, (short)0, c, false, false);
}

// max-reduce across each 16-lane group; result replicated to all 16 lanes.
// DPP16 butterfly (VALU only, no DS traffic): after quad xor1+xor2 all lanes
// in a quad are equal, so half-mirror == xor4 and row-mirror == xor8.
static __device__ __forceinline__ float max_bcast16(float v) {
#if __has_builtin(__builtin_amdgcn_update_dpp)
  int u, t;
  u = __builtin_bit_cast(int, v);
  t = __builtin_amdgcn_update_dpp(u, u, 0xB1, 0xf, 0xf, true);   // quad_perm 1,0,3,2
  v = fmaxf(v, __builtin_bit_cast(float, t));
  u = __builtin_bit_cast(int, v);
  t = __builtin_amdgcn_update_dpp(u, u, 0x4E, 0xf, 0xf, true);   // quad_perm 2,3,0,1
  v = fmaxf(v, __builtin_bit_cast(float, t));
  u = __builtin_bit_cast(int, v);
  t = __builtin_amdgcn_update_dpp(u, u, 0x141, 0xf, 0xf, true);  // row_half_mirror
  v = fmaxf(v, __builtin_bit_cast(float, t));
  u = __builtin_bit_cast(int, v);
  t = __builtin_amdgcn_update_dpp(u, u, 0x140, 0xf, 0xf, true);  // row_mirror
  v = fmaxf(v, __builtin_bit_cast(float, t));
  return v;
#else
  v = fmaxf(v, __shfl_xor(v, 1, 16));
  v = fmaxf(v, __shfl_xor(v, 2, 16));
  v = fmaxf(v, __shfl_xor(v, 4, 16));
  v = fmaxf(v, __shfl_xor(v, 8, 16));
  return v;
#endif
}

// ---------------------------------------------------------------------------
// Phase 1: q = rope(x @ Wq) * SCALE, k = rope(x @ Wk), v = x @ Wv
// q,k row-major bf16 (B,T,32); v transposed bf16 (B,32,T).
// ---------------------------------------------------------------------------
__global__ __launch_bounds__(256) void qkv_rope_kernel(
    const float* __restrict__ x,  const float* __restrict__ Wq,
    const float* __restrict__ Wk, const float* __restrict__ Wv,
    __bf16* __restrict__ qo, __bf16* __restrict__ ko, __bf16* __restrict__ vTo)
{
  int tid = blockIdx.x * blockDim.x + threadIdx.x;   // [0, B*T*16)
  int j   = tid & 15;
  int t   = (tid >> 4) & (TSEQ - 1);
  int b   = tid >> 16;
  int col = 2 * j;

  const float* xr = x + ((size_t)(b * TSEQ + t)) * NEMB;
  float q1 = 0.f, q2 = 0.f, k1 = 0.f, k2 = 0.f, v1 = 0.f, v2 = 0.f;
#pragma unroll 8
  for (int c = 0; c < NEMB; ++c) {
    float xv = xr[c];
    q1 = fmaf(xv, Wq[c * HS + col],     q1);
    q2 = fmaf(xv, Wq[c * HS + col + 1], q2);
    k1 = fmaf(xv, Wk[c * HS + col],     k1);
    k2 = fmaf(xv, Wk[c * HS + col + 1], k2);
    v1 = fmaf(xv, Wv[c * HS + col],     v1);
    v2 = fmaf(xv, Wv[c * HS + col + 1], v2);
  }

  float theta = __expf(-(float)col * (9.210340371976184f / 32.0f));
  float sn, cs;
  __sincosf(theta * (float)t, &sn, &cs);

  size_t ro = ((size_t)(b * TSEQ + t)) * HS + col;
  qo[ro]     = (__bf16)((q1 * cs - q2 * sn) * SCALE);
  qo[ro + 1] = (__bf16)((q1 * sn + q2 * cs) * SCALE);
  ko[ro]     = (__bf16)(k1 * cs - k2 * sn);
  ko[ro + 1] = (__bf16)(k1 * sn + k2 * cs);
  vTo[((size_t)b * HS + col)     * TSEQ + t] = (__bf16)v1;
  vTo[((size_t)b * HS + col + 1) * TSEQ + t] = (__bf16)v2;
}

// ---------------------------------------------------------------------------
// One 64-key flash-attention chunk (4 QK tiles, 4 PV tiles, 2 rowsum tiles).
// ---------------------------------------------------------------------------
template <bool MASK>
static __device__ __forceinline__ void attn_chunk(
    const __bf16* __restrict__ kb, const __bf16* __restrict__ vb,
    __bf16* pslot /* 16 x 64 */, v16bf aq, v16bf ones,
    int kv0, int q0, int ln, int hl,
    float m[8], float l[8], v8f& o0, v8f& o1)
{
  v16bf bk[4];
#pragma unroll
  for (int t = 0; t < 4; ++t)
    bk[t] = *(const v16bf*)(kb + (size_t)(kv0 + 16 * t + ln) * HS + hl * 16);

  v8f z{};
  v8f s[4];
#pragma unroll
  for (int t = 0; t < 4; ++t) s[t] = wmma_bf16(aq, bk[t], z);

#pragma unroll
  for (int r = 0; r < 8; ++r) {
    int qi = q0 + r + 8 * hl;          // C layout: M = r + 8*(lane/16)
    float e[4];
#pragma unroll
    for (int t = 0; t < 4; ++t) {
      e[t] = s[t][r];
      if (MASK) e[t] = (kv0 + 16 * t + ln <= qi) ? e[t] : -1e30f;
    }
    float rm = fmaxf(fmaxf(e[0], e[1]), fmaxf(e[2], e[3]));
    rm = max_bcast16(rm);

    float mn = fmaxf(m[r], rm);
    float cf = __expf(m[r] - mn);
    m[r] = mn;
#pragma unroll
    for (int t = 0; t < 4; ++t) {
      float p = __expf(e[t] - mn);
      pslot[(size_t)(r + 8 * hl) * 64 + ln + 16 * t] = (__bf16)p;
    }
    l[r]  *= cf;
    o0[r] *= cf;
    o1[r] *= cf;
  }

  // same-wave LDS store->load ordering
  asm volatile("s_wait_dscnt 0" ::: "memory");

  v16bf ap0 = *(const v16bf*)(pslot + (size_t)ln * 64 + hl * 16);        // K 0..31
  v16bf ap1 = *(const v16bf*)(pslot + (size_t)ln * 64 + 32 + hl * 16);   // K 32..63

  // row sums of P via the matrix pipe: P(16x64) x ones(64x16)
  v8f ls = wmma_bf16(ap1, ones, wmma_bf16(ap0, ones, z));
#pragma unroll
  for (int r = 0; r < 8; ++r) l[r] += ls[r];

  v16bf bv00 = *(const v16bf*)(vb + (size_t)ln        * TSEQ + kv0 + hl * 16);
  v16bf bv01 = *(const v16bf*)(vb + (size_t)ln        * TSEQ + kv0 + 32 + hl * 16);
  v16bf bv10 = *(const v16bf*)(vb + (size_t)(ln + 16) * TSEQ + kv0 + hl * 16);
  v16bf bv11 = *(const v16bf*)(vb + (size_t)(ln + 16) * TSEQ + kv0 + 32 + hl * 16);
  o0 = wmma_bf16(ap0, bv00, o0);
  o0 = wmma_bf16(ap1, bv01, o0);
  o1 = wmma_bf16(ap0, bv10, o1);
  o1 = wmma_bf16(ap1, bv11, o1);
}

// ---------------------------------------------------------------------------
// Phase 2: flash attention. 1 wave per 16-query tile, 4 waves/block,
// kv in chunks of 64; mask-free bulk loop + exactly one masked diagonal chunk.
// ---------------------------------------------------------------------------
__global__ __launch_bounds__(128) void flash_attn_kernel(
    const __bf16* __restrict__ q, const __bf16* __restrict__ k,
    const __bf16* __restrict__ vT, float* __restrict__ out)
{
  __shared__ __bf16 tileP[4][16][64];   // per-wave P staging, 8KB total

  int wave = threadIdx.x >> 5;
  int lane = threadIdx.x & 31;
  int hl   = lane >> 4;
  int ln   = lane & 15;
  int b    = blockIdx.y;
  int q0   = (blockIdx.x * 4 + wave) * 16;

  const __bf16* qb = q  + (size_t)b * TSEQ * HS;
  const __bf16* kb = k  + (size_t)b * TSEQ * HS;
  const __bf16* vb = vT + (size_t)b * HS * TSEQ;
  __bf16* pslot = &tileP[wave][0][0];

  v16bf aq = *(const v16bf*)(qb + (size_t)(q0 + ln) * HS + hl * 16);

  __bf16 one = (__bf16)1.0f;
  v16bf ones;
#pragma unroll
  for (int i = 0; i < 16; ++i) ones[i] = one;

  float m[8], l[8];
  v8f o0, o1;
#pragma unroll
  for (int r = 0; r < 8; ++r) { m[r] = -1e30f; l[r] = 0.f; o0[r] = 0.f; o1[r] = 0.f; }

  int kv0 = 0;
  for (; kv0 + 64 <= q0; kv0 += 64)        // all keys strictly below the tile
    attn_chunk<false>(kb, vb, pslot, aq, ones, kv0, q0, ln, hl, m, l, o0, o1);
  for (; kv0 < q0 + 16; kv0 += 64)         // exactly one diagonal chunk
    attn_chunk<true>(kb, vb, pslot, aq, ones, kv0, q0, ln, hl, m, l, o0, o1);

  float* ob = out + ((size_t)b * TSEQ + q0) * HS;
#pragma unroll
  for (int r = 0; r < 8; ++r) {
    float inv = 1.0f / l[r];
    ob[(size_t)(r + 8 * hl) * HS + ln]      = o0[r] * inv;
    ob[(size_t)(r + 8 * hl) * HS + ln + 16] = o1[r] * inv;
  }
}

// ---------------------------------------------------------------------------
extern "C" void kernel_launch(void* const* d_in, const int* in_sizes, int n_in,
                              void* d_out, int out_size, void* d_ws, size_t ws_size,
                              hipStream_t stream) {
  const float* x  = (const float*)d_in[0];
  const float* Wq = (const float*)d_in[1];
  const float* Wk = (const float*)d_in[2];
  const float* Wv = (const float*)d_in[3];
  float* out = (float*)d_out;

  __bf16* qws = (__bf16*)d_ws;                       // 2MB
  __bf16* kws = qws + (size_t)BATCH * TSEQ * HS;     // 2MB
  __bf16* vws = kws + (size_t)BATCH * TSEQ * HS;     // 2MB (transposed V)

  int total = BATCH * TSEQ * 16;
  qkv_rope_kernel<<<total / 256, 256, 0, stream>>>(x, Wq, Wk, Wv, qws, kws, vws);

  dim3 grid(TSEQ / 64, BATCH);   // 4 waves/block, 16 q-rows per wave
  flash_attn_kernel<<<grid, 128, 0, stream>>>(qws, kws, vws, out);
}
